// LayeredGatedGCN_3908420240097
// MI455X (gfx1250) — matrix operand order repaired and checked by reading, not compiled, not run on hardware
//
#include <hip/hip_runtime.h>
#include <hip/hip_bf16.h>

#define D     128
#define ASTR  132   // bf16 LDS row stride: 264B = 66 dwords -> conflict-free frags, 8B-aligned packed stores
#define SSTR  132   // f32 staging row stride

typedef __attribute__((ext_vector_type(16))) __bf16 v16bf;
typedef __attribute__((ext_vector_type(8)))  float  v8f;

union Pack4 { __bf16 h[4]; unsigned long long u64; };

__device__ __forceinline__ int laneid() { return threadIdx.x & 31; }
__device__ __forceinline__ int waveid() { return threadIdx.x >> 5; }

__device__ __forceinline__ float wave_sum(float x) {
#pragma unroll
  for (int off = 16; off > 0; off >>= 1) x += __shfl_xor(x, off, 32);
  return x;
}

// Convert a float4 to 4 packed bf16 and store 8B to LDS (address 8B-aligned).
__device__ __forceinline__ void pack_store(__bf16* dst, float4 v) {
  Pack4 pk;
  pk.h[0] = (__bf16)v.x; pk.h[1] = (__bf16)v.y;
  pk.h[2] = (__bf16)v.z; pk.h[3] = (__bf16)v.w;
  *(unsigned long long*)dst = pk.u64;
}

// A fragment (16 rows x 32 K, bf16). Layout per ISA 7.12.2: lanes 0-15/16-31
// hold M=0..15; VGPR v holds K = kb*32 + hi*8 + (v<4 ? 2v : 8+2v) .. +1.
__device__ __forceinline__ v16bf load_afrag(const __bf16* strip, int kb) {
  const int L  = laneid();
  const int hi = L >> 4;
  const __bf16* rowp = strip + (L & 15) * ASTR + kb * 32 + hi * 8;
  v16bf a;
#pragma unroll
  for (int v = 0; v < 8; ++v) {
    const int K = (v < 4) ? 2 * v : 8 + 2 * v;
    a[2 * v]     = rowp[K];
    a[2 * v + 1] = rowp[K + 1];
  }
  return a;
}

// B fragment (32 K x 16 N) for y = x @ W^T: B[k][f] = W[f][k]; W in LDS as [f][d].
// Same K order as A: VGPR v holds K = kb*32 + hi*8 + (v<4 ? 2v : 8+2v) .. +1; N=nt*16+(L&15).
__device__ __forceinline__ v16bf load_bfrag(const __bf16* wlds, int ntile, int kb) {
  const int L  = laneid();
  const int hi = L >> 4;
  const __bf16* rowp = wlds + (ntile * 16 + (L & 15)) * ASTR + kb * 32 + hi * 8;
  v16bf b;
#pragma unroll
  for (int v = 0; v < 8; ++v) {
    b[2 * v]     = rowp[(v < 4) ? 2 * v : 8 + 2 * v];
    b[2 * v + 1] = rowp[((v < 4) ? 2 * v : 8 + 2 * v) + 1];
  }
  return b;
}

// ---------------------------------------------------------------------------
// Kernel 1: 5 node-side linears (A1,A2,A3,B2,B3) for one 128-node tile.
// Output arrays are row-padded to a multiple of 128, so ALL stores are
// unguarded (padding rows are never read downstream). Also zeroes this
// tile's scatter accumulators.
// ---------------------------------------------------------------------------
__global__ void node_linear_kernel(const float* __restrict__ h,
                                   const float* __restrict__ Wl,   // [6][D][D]
                                   const float* __restrict__ bl,   // [6][D]
                                   float* __restrict__ A1h, float* __restrict__ A2h,
                                   float* __restrict__ A3h, float* __restrict__ B2h,
                                   float* __restrict__ B3h,
                                   float* __restrict__ hacc1, float* __restrict__ hacc2,
                                   int Nn) {
  extern __shared__ char smem[];
  __bf16* hbf = (__bf16*)smem;                            // 128 x ASTR
  __bf16* wbf = (__bf16*)(smem + 128 * ASTR * 2);         // 128 x ASTR
  const int tid = threadIdx.x;
  const int m0  = blockIdx.x * 128;
  const bool full = (m0 + 128 <= Nn);

  const float4 z4 = make_float4(0.f, 0.f, 0.f, 0.f);
  for (int idx = tid; idx < 128 * 32; idx += 256) {       // quads: m = idx>>5, q = idx&31
    const int m = idx >> 5, q = idx & 31;
    const int gm = m0 + m;
    float4 v = (full || gm < Nn) ? ((const float4*)(h + (size_t)gm * D))[q] : z4;
    pack_store(hbf + m * ASTR + q * 4, v);
    ((float4*)(hacc1 + (size_t)gm * D))[q] = z4;          // padded arrays: no guard
    ((float4*)(hacc2 + (size_t)gm * D))[q] = z4;
  }
  __syncthreads();

  const int w = waveid(), L = laneid(), hi = L >> 4;
  v16bf afr[4];
#pragma unroll
  for (int kb = 0; kb < 4; ++kb) afr[kb] = load_afrag(hbf + w * 16 * ASTR, kb);

  const int OMAP[5] = {0, 1, 2, 4, 5};
  float* outs[5] = {A1h, A2h, A3h, B2h, B3h};
  const int mrow = m0 + w * 16 + hi * 8;                  // first of the 8 rows this lane stores

  for (int oo = 0; oo < 5; ++oo) {
    const int o = OMAP[oo];
    __syncthreads();
    for (int idx = tid; idx < 128 * 32; idx += 256) {
      const int m = idx >> 5, q = idx & 31;
      pack_store(wbf + m * ASTR + q * 4,
                 ((const float4*)(Wl + (size_t)o * D * D + (size_t)m * D))[q]);
    }
    // Warm L2/L1 for the next weight matrix while this one is consumed.
    if (oo + 1 < 5) {
      const float* nw = Wl + (size_t)OMAP[oo + 1] * D * D + tid * 64;
      __builtin_prefetch(nw, 0, 1);
      __builtin_prefetch(nw + 32, 0, 1);
    }
    __syncthreads();

    float* outp = outs[oo];
#pragma unroll
    for (int nt = 0; nt < 8; ++nt) {                      // fully unrolled: no movrel, static offsets
      const int f = nt * 16 + (L & 15);
      const float bias = bl[o * D + f];
      v8f c;
#pragma unroll
      for (int j = 0; j < 8; ++j) c[j] = bias;
#pragma unroll
      for (int kb = 0; kb < 4; ++kb) {
        v16bf bfr = load_bfrag(wbf, nt, kb);
        c = __builtin_amdgcn_wmma_f32_16x16x32_bf16(false, afr[kb], false, bfr,
                                                    (short)0, c, false, false);
      }
      float* basep = outp + (size_t)mrow * D + f;         // j*D folds into store offsets
#pragma unroll
      for (int j = 0; j < 8; ++j) basep[j * D] = c[j];
    }
  }
}

// ---------------------------------------------------------------------------
// Kernel 2: fused edge pipeline for one 128-edge tile.
// ---------------------------------------------------------------------------
__global__ void edge_kernel(const float* __restrict__ e_in,
                            const float* __restrict__ Wb1,   // W[3]
                            const float* __restrict__ bb1,   // b[3]
                            const float* __restrict__ B2h, const float* __restrict__ B3h,
                            const float* __restrict__ A2h, const float* __restrict__ A3h,
                            const int* __restrict__ rows, const int* __restrict__ cols,
                            const float* __restrict__ lg1, const float* __restrict__ lb1,
                            float* __restrict__ e_out,
                            float* __restrict__ hacc1, float* __restrict__ hacc2,
                            int Ee) {
  extern __shared__ char smem[];
  __bf16* ebf = (__bf16*)smem;                              // 128 x ASTR bf16
  __bf16* wbf = (__bf16*)(smem + 128 * ASTR * 2);           // 128 x ASTR bf16
  float*  sb  = (float*)(smem + 2 * 128 * ASTR * 2);        // 128 x SSTR f32

  const int tid = threadIdx.x;
  const int i0  = blockIdx.x * 128;
  const bool full = (i0 + 128 <= Ee);

  // Prefetch the six L2-resident gather rows for this tile's edges (hints only).
  if (tid < 128) {
    const int gi = i0 + tid;
    if (full || gi < Ee) {
      const int r = rows[gi], c = cols[gi];
      __builtin_prefetch(B2h + (size_t)r * D, 0, 2);
      __builtin_prefetch(B3h + (size_t)c * D, 0, 2);
      __builtin_prefetch(B2h + (size_t)c * D, 0, 2);
      __builtin_prefetch(B3h + (size_t)r * D, 0, 2);
      __builtin_prefetch(A2h + (size_t)r * D, 0, 2);
      __builtin_prefetch(A3h + (size_t)c * D, 0, 2);
    }
  }

  const float4 z4 = make_float4(0.f, 0.f, 0.f, 0.f);
  for (int idx = tid; idx < 128 * 32; idx += 256) {
    const int m = idx >> 5, q = idx & 31;
    const int gi = i0 + m;
    float4 ev = (full || gi < Ee) ? ((const float4*)(e_in + (size_t)gi * D))[q] : z4;
    pack_store(ebf + m * ASTR + q * 4, ev);
    pack_store(wbf + m * ASTR + q * 4, ((const float4*)(Wb1 + (size_t)m * D))[q]);
  }
  __syncthreads();

  const int w = waveid(), L = laneid(), hi = L >> 4;
  {
    v16bf afr[4];
#pragma unroll
    for (int kb = 0; kb < 4; ++kb) afr[kb] = load_afrag(ebf + w * 16 * ASTR, kb);
#pragma unroll
    for (int nt = 0; nt < 8; ++nt) {
      const int f = nt * 16 + (L & 15);
      const float bias = bb1[f];
      v8f c;
#pragma unroll
      for (int j = 0; j < 8; ++j) c[j] = bias;
#pragma unroll
      for (int kb = 0; kb < 4; ++kb) {
        v16bf bfr = load_bfrag(wbf, nt, kb);
        c = __builtin_amdgcn_wmma_f32_16x16x32_bf16(false, afr[kb], false, bfr,
                                                    (short)0, c, false, false);
      }
#pragma unroll
      for (int j = 0; j < 8; ++j)
        sb[(w * 16 + hi * 8 + j) * SSTR + f] = c[j];
    }
  }
  __syncthreads();

  // Row phase: one wave per edge row, 4 features per lane (d = L + 32q).
  for (int it = 0; it < 16; ++it) {
    const int rloc = it * 8 + w;
    const int i = i0 + rloc;
    if (!full && i >= Ee) continue;                       // wave-uniform
    const int r  = rows[i];
    const int cI = cols[i];

    float b1[4], b2r[4], b3c[4], b2c[4], b3r[4], ein[4], g[4], be[4];
#pragma unroll
    for (int q = 0; q < 4; ++q) {
      const int d = L + 32 * q;
      b1[q]  = sb[rloc * SSTR + d];
      b2r[q] = B2h[(size_t)r  * D + d];
      b3c[q] = B3h[(size_t)cI * D + d];
      b2c[q] = B2h[(size_t)cI * D + d];
      b3r[q] = B3h[(size_t)r  * D + d];
      ein[q] = e_in[(size_t)i * D + d];
      g[q]   = lg1[d];
      be[q]  = lb1[d];
    }

    float xji[4], xik[4], s1 = 0.f, q1 = 0.f, s2 = 0.f, q2 = 0.f;
#pragma unroll
    for (int q = 0; q < 4; ++q) {
      float a = fmaxf(b1[q] + b2r[q] + b3c[q], 0.f); xji[q] = a; s1 += a; q1 += a * a;
      float b = fmaxf(b1[q] + b2c[q] + b3r[q], 0.f); xik[q] = b; s2 += b; q2 += b * b;
    }
    s1 = wave_sum(s1); q1 = wave_sum(q1);
    s2 = wave_sum(s2); q2 = wave_sum(q2);
    const float mu1 = s1 * (1.0f / D), va1 = q1 * (1.0f / D) - mu1 * mu1;
    const float mu2 = s2 * (1.0f / D), va2 = q2 * (1.0f / D) - mu2 * mu2;
    const float rs1 = rsqrtf(va1 + 1e-5f), rs2 = rsqrtf(va2 + 1e-5f);

    float eji[4], eik[4], sg1[4], sg2[4], ss1 = 0.f, ss2 = 0.f;
#pragma unroll
    for (int q = 0; q < 4; ++q) {
      eji[q] = ein[q] + (xji[q] - mu1) * rs1 * g[q] + be[q];
      eik[q] = ein[q] + (xik[q] - mu2) * rs2 * g[q] + be[q];
      sg1[q] = 1.0f / (1.0f + __expf(-eji[q])); ss1 += sg1[q];
      sg2[q] = 1.0f / (1.0f + __expf(-eik[q])); ss2 += sg2[q];
    }
    ss1 = wave_sum(ss1) + 1e-6f;
    ss2 = wave_sum(ss2) + 1e-6f;
    const float is1 = 1.0f / ss1, is2 = 1.0f / ss2;

#pragma unroll
    for (int q = 0; q < 4; ++q) {
      const int d = L + 32 * q;
      e_out[(size_t)i * D + d] = eji[q];
      const float mji = A2h[(size_t)r  * D + d] * sg1[q] * is1;
      unsafeAtomicAdd(&hacc1[(size_t)cI * D + d], mji);
      const float mik = A3h[(size_t)cI * D + d] * sg2[q] * is2;
      unsafeAtomicAdd(&hacc2[(size_t)r  * D + d], mik);
    }
  }
}

// ---------------------------------------------------------------------------
// Kernel 3: h_out = h_in + LN(relu(A1h + hacc1 + hacc2)); one wave per node.
// ---------------------------------------------------------------------------
__global__ void node_finalize_kernel(const float* __restrict__ h_in,
                                     const float* __restrict__ A1h,
                                     const float* __restrict__ hacc1,
                                     const float* __restrict__ hacc2,
                                     const float* __restrict__ lg0,
                                     const float* __restrict__ lb0,
                                     float* __restrict__ h_out, int Nn) {
  const int w = threadIdx.x >> 5, L = threadIdx.x & 31;
  const int i = blockIdx.x * 8 + w;
  if (i >= Nn) return;                                    // wave-uniform
  float x[4], s = 0.f, q = 0.f;
#pragma unroll
  for (int qq = 0; qq < 4; ++qq) {
    const int d = L + 32 * qq;
    float v = A1h[(size_t)i * D + d] + hacc1[(size_t)i * D + d] + hacc2[(size_t)i * D + d];
    v = fmaxf(v, 0.f);
    x[qq] = v; s += v; q += v * v;
  }
  s = wave_sum(s); q = wave_sum(q);
  const float mu = s * (1.0f / D), va = q * (1.0f / D) - mu * mu;
  const float rs = rsqrtf(va + 1e-5f);
#pragma unroll
  for (int qq = 0; qq < 4; ++qq) {
    const int d = L + 32 * qq;
    h_out[(size_t)i * D + d] = h_in[(size_t)i * D + d] + (x[qq] - mu) * rs * lg0[d] + lb0[d];
  }
}

// ---------------------------------------------------------------------------
extern "C" void kernel_launch(void* const* d_in, const int* in_sizes, int n_in,
                              void* d_out, int out_size, void* d_ws, size_t ws_size,
                              hipStream_t stream) {
  (void)n_in; (void)out_size; (void)ws_size;
  const float* h_input = (const float*)d_in[0];
  const float* e_input = (const float*)d_in[1];
  const int*   eidx    = (const int*)d_in[2];
  const float* Ws      = (const float*)d_in[3];
  const float* bs      = (const float*)d_in[4];
  const float* lng     = (const float*)d_in[5];
  const float* lnb     = (const float*)d_in[6];

  const int N  = in_sizes[0] / D;
  const int E  = in_sizes[1] / D;
  const int NL = in_sizes[3] / (6 * D * D);
  const int* rows = eidx;
  const int* cols = eidx + E;

  float* out_h = (float*)d_out;
  float* out_e = out_h + (size_t)N * D;

  const int nodeTiles = (N + 127) / 128;
  const int Npad = nodeTiles * 128;        // intermediate node arrays padded -> unguarded GEMM stores

  float* p = (float*)d_ws;
  float* eA    = p; p += (size_t)E * D;
  float* hA    = p; p += (size_t)N * D;
  float* A1h   = p; p += (size_t)Npad * D;
  float* A2h   = p; p += (size_t)Npad * D;
  float* A3h   = p; p += (size_t)Npad * D;
  float* B2h   = p; p += (size_t)Npad * D;
  float* B3h   = p; p += (size_t)Npad * D;
  float* hacc1 = p; p += (size_t)Npad * D;
  float* hacc2 = p; p += (size_t)Npad * D;

  const int edgeTiles = (E + 127) / 128;
  const int finBlocks = (N + 7) / 8;
  const size_t lds_node = (size_t)2 * 128 * ASTR * 2;                           // ~66 KB
  const size_t lds_edge = (size_t)2 * 128 * ASTR * 2 + (size_t)128 * SSTR * 4;  // ~132 KB

  hipFuncSetAttribute((const void*)node_linear_kernel,
                      hipFuncAttributeMaxDynamicSharedMemorySize, (int)lds_node);
  hipFuncSetAttribute((const void*)edge_kernel,
                      hipFuncAttributeMaxDynamicSharedMemorySize, (int)lds_edge);

  const float* hcur = h_input;
  const float* ecur = e_input;
  for (int l = 0; l < NL; ++l) {
    const float* Wl  = Ws  + (size_t)l * 6 * D * D;
    const float* bl  = bs  + (size_t)l * 6 * D;
    const float* lg0 = lng + (size_t)l * 2 * D;
    const float* lg1 = lg0 + D;
    const float* lb0 = lnb + (size_t)l * 2 * D;
    const float* lb1 = lb0 + D;
    float* hout = (l % 2 == 0) ? hA : out_h;   // NL=8 even -> layer 7 lands in d_out
    float* eout = (l % 2 == 0) ? eA : out_e;

    node_linear_kernel<<<nodeTiles, 256, lds_node, stream>>>(
        hcur, Wl, bl, A1h, A2h, A3h, B2h, B3h, hacc1, hacc2, N);
    edge_kernel<<<edgeTiles, 256, lds_edge, stream>>>(
        ecur, Wl + 3 * D * D, bl + 3 * D, B2h, B3h, A2h, A3h,
        rows, cols, lg1, lb1, eout, hacc1, hacc2, E);
    node_finalize_kernel<<<finBlocks, 256, 0, stream>>>(
        hcur, A1h, hacc1, hacc2, lg0, lb0, hout, N);

    hcur = hout; ecur = eout;
  }
}
